// GNN_90159953478165
// MI455X (gfx1250) — compile-verified, matrix operand order checked by reading, not run on hardware
//
#include <hip/hip_runtime.h>
#include <hip/hip_bf16.h>

typedef __attribute__((ext_vector_type(16))) _Float16 v16h;
typedef __attribute__((ext_vector_type(8)))  _Float16 v8h;
typedef __attribute__((ext_vector_type(8)))  float    v8f;

#define N_NODES 50000
#define N_EDGES 800000
#define N_GRAPHS 8

// ---------------- helpers ----------------
__device__ __forceinline__ float lrelu02(float x) { return x > 0.f ? x : 0.2f * x; }
__device__ __forceinline__ float eluf(float x)    { return x > 0.f ? x : (expf(x) - 1.f); }

// monotone float<->uint encoding for atomicMax on floats
__device__ __forceinline__ unsigned encf(float f) {
    unsigned u = __float_as_uint(f);
    return (u & 0x80000000u) ? ~u : (u | 0x80000000u);
}
__device__ __forceinline__ float decf(unsigned e) {
    unsigned u = (e & 0x80000000u) ? (e ^ 0x80000000u) : ~e;
    return __uint_as_float(u);
}
#define ENC_NEG_INF 0x007FFFFFu  // encf(-inf)

__device__ __forceinline__ float waveRedSum(float v) {
    v += __shfl_xor(v, 16, 32);
    v += __shfl_xor(v, 8, 32);
    v += __shfl_xor(v, 4, 32);
    v += __shfl_xor(v, 2, 32);
    v += __shfl_xor(v, 1, 32);
    return v;
}

// ---------------- fills ----------------
__global__ void k_fill_f32(float* p, float v, int n) {
    int i = blockIdx.x * blockDim.x + threadIdx.x;
    if (i < n) p[i] = v;
}
__global__ void k_fill_u32(unsigned* p, unsigned v, int n) {
    int i = blockIdx.x * blockDim.x + threadIdx.x;
    if (i < n) p[i] = v;
}

// ---------------- weight prep: WT[n*K+k] = (f16) W[k*N+n] ----------------
__global__ void k_prep_wT(const float* __restrict__ W, _Float16* __restrict__ WT, int N, int K) {
    int i = blockIdx.x * blockDim.x + threadIdx.x;
    if (i >= N * K) return;
    int n = i / K, k = i - n * K;
    WT[i] = (_Float16)W[k * N + n];
}

// ---------------- WMMA GEMM: C[M,N] = A[M,K] @ W[K,N] + bias[N] ----------------
// One wave per 16(M) x 64(N) tile: A fragment reused across 4 back-to-back WMMAs.
// Weights pre-transposed/converted to f16 (WT[n*K+k]) -> B fragment = 2x b128 loads.
// K is a template constant (multiple of 32) so the chunk loop fully unrolls.
template <int K>
__global__ void k_gemm_wmma(const float* __restrict__ A, const _Float16* __restrict__ WT,
                            const float* __restrict__ bias, float* __restrict__ C,
                            int lda, int ldc) {
    const int lane  = threadIdx.x;      // 0..31
    const int half  = lane >> 4;        // 0/1
    const int l16   = lane & 15;
    const int row   = blockIdx.x * 16 + l16;   // A row this lane feeds
    const int ncol0 = blockIdx.y * 64;         // base output column of this wave's tile

    v8f acc[4] = {};
    const float* arow = A + row * lda;
#pragma unroll
    for (int k0 = 0; k0 < K; k0 += 32) {
        // ---- A fragment: elements 0..7 -> K = k0+half*8+0..7 ; 8..15 -> +16
        const float* ab = arow + k0 + half * 8;
        float4 x0 = *(const float4*)(ab);
        float4 x1 = *(const float4*)(ab + 4);
        float4 x2 = *(const float4*)(ab + 16);
        float4 x3 = *(const float4*)(ab + 20);
        v16h af;
        af[0] = (_Float16)x0.x;  af[1] = (_Float16)x0.y;  af[2]  = (_Float16)x0.z;  af[3]  = (_Float16)x0.w;
        af[4] = (_Float16)x1.x;  af[5] = (_Float16)x1.y;  af[6]  = (_Float16)x1.z;  af[7]  = (_Float16)x1.w;
        af[8] = (_Float16)x2.x;  af[9] = (_Float16)x2.y;  af[10] = (_Float16)x2.z;  af[11] = (_Float16)x2.w;
        af[12]= (_Float16)x3.x;  af[13]= (_Float16)x3.y;  af[14] = (_Float16)x3.z;  af[15] = (_Float16)x3.w;
        // ---- 4 B fragments, WMMA back-to-back on the same A
#pragma unroll
        for (int t = 0; t < 4; ++t) {
            const int col = ncol0 + t * 16 + l16;
            const _Float16* wb = WT + col * K + k0 + half * 8;
            v8h b0 = *(const v8h*)(wb);
            v8h b1 = *(const v8h*)(wb + 16);
            v16h bf;
#pragma unroll
            for (int e = 0; e < 8; ++e) { bf[e] = b0[e]; bf[e + 8] = b1[e]; }
            acc[t] = __builtin_amdgcn_wmma_f32_16x16x32_f16(
                /*neg_a=*/false, af, /*neg_b=*/false, bf,
                /*c_mod=*/(short)0, acc[t], /*reuse_a=*/false, /*reuse_b=*/false);
        }
    }
#pragma unroll
    for (int t = 0; t < 4; ++t) {
        const int col = ncol0 + t * 16 + l16;
        const float bc = bias[col];
#pragma unroll
        for (int r = 0; r < 8; ++r) {
            int m = blockIdx.x * 16 + r + half * 8;   // C/D layout: VGPR r -> row r + 8*half
            C[m * ldc + col] = acc[t][r] + bc;
        }
    }
}

// ---------------- elu + per-row LayerNorm over 64 channels (wave per row) ----------------
__global__ void k_elu_ln64(const float* __restrict__ X, float* __restrict__ Y,
                           const float* __restrict__ w, const float* __restrict__ b, int rows) {
    int wid  = blockIdx.x * (blockDim.x / 32) + (threadIdx.x >> 5);
    int lane = threadIdx.x & 31;
    if (wid >= rows) return;
    const float* xr = X + wid * 64;
    float v0 = eluf(xr[lane]), v1 = eluf(xr[lane + 32]);
    float mu = waveRedSum(v0 + v1) * (1.f / 64.f);
    float d0 = v0 - mu, d1 = v1 - mu;
    float var = waveRedSum(d0 * d0 + d1 * d1) * (1.f / 64.f);
    float rs = rsqrtf(var + 1e-5f);
    float* yr = Y + wid * 64;
    yr[lane]      = d0 * rs * w[lane]      + b[lane];
    yr[lane + 32] = d1 * rs * w[lane + 32] + b[lane + 32];
}

// ---------------- attention logits + segment max (wave per edge-head) ----------------
__global__ void k_edge_alpha(const int* __restrict__ ei, const float* __restrict__ ea,
                             const float* __restrict__ XL, const float* __restrict__ XR,
                             const float* __restrict__ we, const float* __restrict__ att,
                             float* __restrict__ alpha, unsigned* __restrict__ amaxe,
                             int E, int H, int ldx) {
    int w    = blockIdx.x * (blockDim.x / 32) + (threadIdx.x >> 5);
    int lane = threadIdx.x & 31;
    if (w >= E * H) return;
    int e = w / H, h = w - e * H;
    int src = ei[e], dst = ei[E + e];
    const int ldw = H * 64;
    const float* eav = ea + e * 16;
    float part = 0.f;
#pragma unroll
    for (int j = 0; j < 2; ++j) {
        int c   = lane + j * 32;
        int col = h * 64 + c;
        float ec = 0.f;
#pragma unroll
        for (int k = 0; k < 16; ++k) ec += eav[k] * we[k * ldw + col];   // edge embed on the fly
        float m = XL[src * ldx + col] + XR[dst * ldx + col] + ec;
        part += lrelu02(m) * att[col];
    }
    part = waveRedSum(part);
    if (lane == 0) {
        alpha[w] = part;
        atomicMax(&amaxe[dst * H + h], encf(part));
    }
}

__global__ void k_fix_amax(const unsigned* __restrict__ amaxe, float* __restrict__ amaxf, int n) {
    int i = blockIdx.x * blockDim.x + threadIdx.x;
    if (i >= n) return;
    float d = decf(amaxe[i]);
    amaxf[i] = __builtin_isfinite(d) ? d : 0.f;
}

__global__ void k_edge_denom(const int* __restrict__ ei, float* __restrict__ alpha,
                             const float* __restrict__ amaxf, float* __restrict__ denom,
                             int E, int H) {
    int w = blockIdx.x * blockDim.x + threadIdx.x;
    if (w >= E * H) return;
    int e = w / H, h = w - e * H;
    int dst = ei[E + e];
    float p = expf(alpha[w] - amaxf[dst * H + h]);
    alpha[w] = p;
    atomicAdd(&denom[dst * H + h], p);
}

// ---------------- weighted scatter: OUT[dst] += XL[src]*alpha  (block per edge) ----------------
__global__ void k_edge_scatter(const int* __restrict__ ei, const float* __restrict__ alpha,
                               const float* __restrict__ denom, const float* __restrict__ XL,
                               float* __restrict__ OUT, int E, int H, int ldx) {
    int e  = blockIdx.x;
    int cf = threadIdx.x;               // 0 .. H*64-1
    int h  = cf >> 6;
    int src = ei[e], dst = ei[E + e];
    float wgt = alpha[e * H + h] / (denom[dst * H + h] + 1e-16f);
    atomicAdd(&OUT[dst * ldx + cf], XL[src * ldx + cf] * wgt);
}

__global__ void k_bias(float* __restrict__ p, const float* __restrict__ b, int n, int ld) {
    int i = blockIdx.x * blockDim.x + threadIdx.x;
    if (i < n) p[i] += b[i & (ld - 1)];
}

// ---------------- graph norm (global mean/var) ----------------
__global__ void k_gn_stats(const float* __restrict__ X, float* __restrict__ stats, int n) {
    __shared__ float ssum[8], ssq[8];
    int i = blockIdx.x * blockDim.x + threadIdx.x;
    float v = (i < n) ? X[i] : 0.f;
    float s = waveRedSum(v);
    float q = waveRedSum(v * v);
    int lane = threadIdx.x & 31, wid = threadIdx.x >> 5;
    if (lane == 0) { ssum[wid] = s; ssq[wid] = q; }
    __syncthreads();
    if (threadIdx.x == 0) {
        float ts = 0.f, tq = 0.f;
        for (int k = 0; k < 8; ++k) { ts += ssum[k]; tq += ssq[k]; }
        atomicAdd(&stats[0], ts);
        atomicAdd(&stats[1], tq);
    }
}

__global__ void k_gn_apply(const float* __restrict__ X, float* __restrict__ Y,
                           const float* __restrict__ stats, const float* __restrict__ w,
                           const float* __restrict__ b, int n, int Cmask, float invCnt) {
    int i = blockIdx.x * blockDim.x + threadIdx.x;
    if (i >= n) return;
    float mu  = stats[0] * invCnt;
    float var = stats[1] * invCnt - mu * mu;
    float rs  = rsqrtf(var + 1e-5f);
    int c = i & Cmask;
    Y[i] = eluf((X[i] - mu) * rs * w[c] + b[c]);
}

// ---------------- output heads ----------------
__global__ void k_heads(const float* __restrict__ H3, const int* __restrict__ batch,
                        const float* __restrict__ pw, const float* __restrict__ pb,
                        float* __restrict__ out, float* __restrict__ gsum,
                        float* __restrict__ gcnt, int rows) {
    int wid  = blockIdx.x * (blockDim.x / 32) + (threadIdx.x >> 5);
    int lane = threadIdx.x & 31;
    if (wid >= rows) return;
    int g = batch[wid];
    const float* hr = H3 + wid * 64;
    float h0 = hr[lane], h1 = hr[lane + 32];
    atomicAdd(&gsum[g * 64 + lane], h0);
    atomicAdd(&gsum[g * 64 + lane + 32], h1);
    float part = waveRedSum(h0 * pw[lane] + h1 * pw[lane + 32]);
    if (lane == 0) {
        out[wid] = part + pb[0];
        atomicAdd(&gcnt[g], 1.f);
    }
}

__global__ void k_hp(const float* __restrict__ gsum, const float* __restrict__ gcnt,
                     const float* __restrict__ hw, const float* __restrict__ hb,
                     float* __restrict__ out) {
    int g = threadIdx.x >> 5, lane = threadIdx.x & 31;
    if (g >= N_GRAPHS) return;
    float cnt = fmaxf(gcnt[g], 1.f);
    float e0 = gsum[g * 64 + lane] / cnt;
    float e1 = gsum[g * 64 + lane + 32] / cnt;
    float part = waveRedSum(e0 * hw[lane] + e1 * hw[lane + 32]);
    if (lane == 0) out[g] = part + hb[0];
}

// ---------------- one GATv2 layer ----------------
template <int K>
static void run_gat_layer(const float* Hin, const int* ei, const float* ea,
                          const _Float16* wlT, const float* bl, const _Float16* wrT, const float* br,
                          const float* we, const float* att, const float* bias,
                          float* XL, float* XR, float* OUT, float* alphaB, float* denom,
                          unsigned* amaxe, float* amaxf, int H, hipStream_t stream) {
    const int NN = N_NODES, EE = N_EDGES;
    const int Nout = H * 64;
    dim3 gg(NN / 16, Nout / 64);
    k_gemm_wmma<K><<<gg, 32, 0, stream>>>(Hin, wlT, bl, XL, K, Nout);
    k_gemm_wmma<K><<<gg, 32, 0, stream>>>(Hin, wrT, br, XR, K, Nout);
    int nh = NN * H;
    k_fill_u32<<<(nh + 255) / 256, 256, 0, stream>>>(amaxe, ENC_NEG_INF, nh);
    k_fill_f32<<<(nh + 255) / 256, 256, 0, stream>>>(denom, 0.f, nh);
    k_fill_f32<<<(NN * Nout + 255) / 256, 256, 0, stream>>>(OUT, 0.f, NN * Nout);
    int EH = EE * H;
    k_edge_alpha<<<EH / 4, 128, 0, stream>>>(ei, ea, XL, XR, we, att, alphaB, amaxe, EE, H, Nout);
    k_fix_amax<<<(nh + 255) / 256, 256, 0, stream>>>(amaxe, amaxf, nh);
    k_edge_denom<<<EH / 256, 256, 0, stream>>>(ei, alphaB, amaxf, denom, EE, H);
    k_edge_scatter<<<EE, Nout, 0, stream>>>(ei, alphaB, denom, XL, OUT, EE, H, Nout);
    k_bias<<<(NN * Nout + 255) / 256, 256, 0, stream>>>(OUT, bias, NN * Nout, Nout);
}

extern "C" void kernel_launch(void* const* d_in, const int* in_sizes, int n_in,
                              void* d_out, int out_size, void* d_ws, size_t ws_size,
                              hipStream_t stream) {
    (void)in_sizes; (void)n_in; (void)out_size; (void)ws_size;
    const float* x     = (const float*)d_in[0];
    const int*   ei    = (const int*)d_in[1];
    const float* ea    = (const float*)d_in[2];
    const int*   batch = (const int*)d_in[3];
    const float* nt_w  = (const float*)d_in[4];  const float* nt_b  = (const float*)d_in[5];
    const float* ln0_w = (const float*)d_in[6];  const float* ln0_b = (const float*)d_in[7];
    const float* c1_wl = (const float*)d_in[8];  const float* c1_bl = (const float*)d_in[9];
    const float* c1_wr = (const float*)d_in[10]; const float* c1_br = (const float*)d_in[11];
    const float* c1_we = (const float*)d_in[12]; const float* c1_att= (const float*)d_in[13];
    const float* c1_b  = (const float*)d_in[14];
    const float* n1_w  = (const float*)d_in[15]; const float* n1_b  = (const float*)d_in[16];
    const float* c2_wl = (const float*)d_in[17]; const float* c2_bl = (const float*)d_in[18];
    const float* c2_wr = (const float*)d_in[19]; const float* c2_br = (const float*)d_in[20];
    const float* c2_we = (const float*)d_in[21]; const float* c2_att= (const float*)d_in[22];
    const float* c2_b  = (const float*)d_in[23];
    const float* n2_w  = (const float*)d_in[24]; const float* n2_b  = (const float*)d_in[25];
    const float* c3_wl = (const float*)d_in[26]; const float* c3_bl = (const float*)d_in[27];
    const float* c3_wr = (const float*)d_in[28]; const float* c3_br = (const float*)d_in[29];
    const float* c3_we = (const float*)d_in[30]; const float* c3_att= (const float*)d_in[31];
    const float* c3_b  = (const float*)d_in[32];
    const float* pof_w = (const float*)d_in[33]; const float* pof_b = (const float*)d_in[34];
    const float* hp_w  = (const float*)d_in[35]; const float* hp_b  = (const float*)d_in[36];

    float* out = (float*)d_out;
    float* ws  = (float*)d_ws;

    const int NN = N_NODES, EE = N_EDGES;
    size_t off = 0;
    float*    A      = ws + off;               off += (size_t)NN * 128;
    float*    Bb     = ws + off;               off += (size_t)NN * 128;
    float*    XL     = ws + off;               off += (size_t)NN * 128;
    float*    XR     = ws + off;               off += (size_t)NN * 128;
    float*    alphaB = ws + off;               off += (size_t)EE * 2;
    float*    denom  = ws + off;               off += (size_t)NN * 2;
    unsigned* amaxe  = (unsigned*)(ws + off);  off += (size_t)NN * 2;
    float*    amaxf  = ws + off;               off += (size_t)NN * 2;
    float*    stats  = ws + off;               off += 2;
    float*    gsum   = ws + off;               off += 64 * N_GRAPHS;   // gcnt follows contiguously
    float*    gcnt   = ws + off;               off += N_GRAPHS;
    off = (off + 3) & ~(size_t)3;              // 16B align for f16 vector loads

    _Float16* hb     = (_Float16*)(ws + off);
    size_t ho = 0;
    _Float16* nt_wT  = hb + ho;  ho += 64 * 32;
    _Float16* c1_wlT = hb + ho;  ho += 128 * 64;
    _Float16* c1_wrT = hb + ho;  ho += 128 * 64;
    _Float16* c2_wlT = hb + ho;  ho += 128 * 128;
    _Float16* c2_wrT = hb + ho;  ho += 128 * 128;
    _Float16* c3_wlT = hb + ho;  ho += 64 * 128;
    _Float16* c3_wrT = hb + ho;  ho += 64 * 128;

    // ---- prep: transpose+convert all GEMM weights to f16 (tiny) ----
    k_prep_wT<<<(64 * 32 + 255) / 256, 256, 0, stream>>>(nt_w, nt_wT, 64, 32);
    k_prep_wT<<<(128 * 64 + 255) / 256, 256, 0, stream>>>(c1_wl, c1_wlT, 128, 64);
    k_prep_wT<<<(128 * 64 + 255) / 256, 256, 0, stream>>>(c1_wr, c1_wrT, 128, 64);
    k_prep_wT<<<(128 * 128 + 255) / 256, 256, 0, stream>>>(c2_wl, c2_wlT, 128, 128);
    k_prep_wT<<<(128 * 128 + 255) / 256, 256, 0, stream>>>(c2_wr, c2_wrT, 128, 128);
    k_prep_wT<<<(64 * 128 + 255) / 256, 256, 0, stream>>>(c3_wl, c3_wlT, 64, 128);
    k_prep_wT<<<(64 * 128 + 255) / 256, 256, 0, stream>>>(c3_wr, c3_wrT, 64, 128);

    // ---- node_transform: h0 = LN(elu(x @ nt_w + nt_b)) ----
    {
        dim3 gg(NN / 16, 1);
        k_gemm_wmma<32><<<gg, 32, 0, stream>>>(x, nt_wT, nt_b, Bb, 32, 64);
        k_elu_ln64<<<NN / 8, 256, 0, stream>>>(Bb, A, ln0_w, ln0_b, NN);
    }

    // ---- layer 1: GATv2 (K=64, H=2) -> graph_norm -> elu ----
    run_gat_layer<64>(A, ei, ea, c1_wlT, c1_bl, c1_wrT, c1_br, c1_we, c1_att, c1_b,
                      XL, XR, Bb, alphaB, denom, amaxe, amaxf, 2, stream);
    k_fill_f32<<<1, 256, 0, stream>>>(stats, 0.f, 2);
    k_gn_stats<<<NN * 128 / 256, 256, 0, stream>>>(Bb, stats, NN * 128);
    k_gn_apply<<<NN * 128 / 256, 256, 0, stream>>>(Bb, A, stats, n1_w, n1_b,
                                                   NN * 128, 127, 1.f / ((float)NN * 128.f));

    // ---- layer 2: GATv2 (K=128, H=2) -> graph_norm -> elu ----
    run_gat_layer<128>(A, ei, ea, c2_wlT, c2_bl, c2_wrT, c2_br, c2_we, c2_att, c2_b,
                       XL, XR, Bb, alphaB, denom, amaxe, amaxf, 2, stream);
    k_fill_f32<<<1, 256, 0, stream>>>(stats, 0.f, 2);
    k_gn_stats<<<NN * 128 / 256, 256, 0, stream>>>(Bb, stats, NN * 128);
    k_gn_apply<<<NN * 128 / 256, 256, 0, stream>>>(Bb, A, stats, n2_w, n2_b,
                                                   NN * 128, 127, 1.f / ((float)NN * 128.f));

    // ---- layer 3: GATv2 (K=128, H=1), no norm ----
    run_gat_layer<128>(A, ei, ea, c3_wlT, c3_bl, c3_wrT, c3_br, c3_we, c3_att, c3_b,
                       XL, XR, Bb, alphaB, denom, amaxe, amaxf, 1, stream);

    // ---- heads: pof per node, mean-pooled graph head ----
    k_fill_f32<<<(64 * N_GRAPHS + N_GRAPHS + 255) / 256, 256, 0, stream>>>(gsum, 0.f,
                                                                           64 * N_GRAPHS + N_GRAPHS);
    k_heads<<<NN / 8, 256, 0, stream>>>(Bb, batch, pof_w, pof_b, out, gsum, gcnt, NN);
    k_hp<<<1, 256, 0, stream>>>(gsum, gcnt, hp_w, hp_b, out + NN);
}
